// Decoder_33921651703843
// MI455X (gfx1250) — compile-verified, matrix operand order checked by reading, not run on hardware
//
#include <hip/hip_runtime.h>
#include <hip/hip_bf16.h>

typedef __attribute__((ext_vector_type(16))) _Float16 v16h;
typedef __attribute__((ext_vector_type(8)))  _Float16 v8h;
typedef __attribute__((ext_vector_type(8)))  float    v8f;

// ---------------------------------------------------------------------------
// Fragment loader: 16-bit A/B fragment per CDNA5 WMMA layout.
// lane 0-15 : row = base+ (lane&15), halves = K[k0+0..7], K[k0+16..23]
// lane 16-31: row = base+ (lane&15), halves = K[k0+8..15], K[k0+24..31]
// Two contiguous b128 loads per fragment.
// ---------------------------------------------------------------------------
__device__ __forceinline__ v16h load_frag(const _Float16* __restrict__ base,
                                          int row, int ld, int k0, int sel) {
  const _Float16* p = base + (size_t)row * ld + k0 + sel * 8;
  v8h lo = *(const v8h*)(p);
  v8h hi = *(const v8h*)(p + 16);
  v16h r;
#pragma unroll
  for (int i = 0; i < 8; ++i) { r[i] = lo[i]; r[8 + i] = hi[i]; }
  return r;
}

// ---------------------------------------------------------------------------
// Generic WMMA GEMM:  C[M,N] = scale * (A(MxK,f16) @ B(NxK,f16)^T) + bias[n]
// One wave (32 threads) computes a 64x64 tile: 4x4 accumulators of 16x16.
// Optional f32 and/or f16 outputs. Batched via blockIdx.z with two-level
// output stride: cofs = (z/zdiv)*sChi + (z%zdiv)*sClo.
// ---------------------------------------------------------------------------
__global__ __launch_bounds__(32)
void wmma_gemm_abt(const _Float16* __restrict__ A, int lda, long sA,
                   const _Float16* __restrict__ B, int ldb, long sB,
                   float* __restrict__ Cf, _Float16* __restrict__ Ch, int ldc,
                   long sChi, long sClo, int zdiv,
                   const float* __restrict__ bias, float scale, int relu,
                   int K) {
  const int lane = threadIdx.x & 31;
  const int sel  = lane >> 4;
  const int lr   = lane & 15;
  const int z    = blockIdx.z;
  A += (long)z * sA;
  B += (long)z * sB;
  const long cofs = (long)(z / zdiv) * sChi + (long)(z % zdiv) * sClo;
  const int m0 = blockIdx.y * 64;
  const int n0 = blockIdx.x * 64;

  v8f acc[4][4] = {};

  for (int k0 = 0; k0 < K; k0 += 32) {
    if (k0 + 32 < K) {  // pull next K slab toward L0/L2 (global_prefetch_b8)
      __builtin_prefetch(A + (size_t)(m0 + lr) * lda + k0 + 32, 0, 1);
      __builtin_prefetch(B + (size_t)(n0 + lr) * ldb + k0 + 32, 0, 1);
    }
    v16h af[4], bf[4];
#pragma unroll
    for (int i = 0; i < 4; ++i) af[i] = load_frag(A, m0 + i * 16 + lr, lda, k0, sel);
#pragma unroll
    for (int j = 0; j < 4; ++j) bf[j] = load_frag(B, n0 + j * 16 + lr, ldb, k0, sel);
#pragma unroll
    for (int i = 0; i < 4; ++i)
#pragma unroll
      for (int j = 0; j < 4; ++j)
        acc[i][j] = __builtin_amdgcn_wmma_f32_16x16x32_f16(
            false, af[i], false, bf[j], (short)0, acc[i][j], false, false);
  }

  // Epilogue: C/D layout — VGPR r holds row m0+i*16+sel*8+r, col n0+j*16+lr.
#pragma unroll
  for (int i = 0; i < 4; ++i) {
#pragma unroll
    for (int r = 0; r < 8; ++r) {
      const int row = m0 + i * 16 + sel * 8 + r;
#pragma unroll
      for (int j = 0; j < 4; ++j) {
        const int col = n0 + j * 16 + lr;
        float v = acc[i][j][r] * scale + (bias ? bias[col] : 0.f);
        if (relu) v = fmaxf(v, 0.f);
        const long idx = cofs + (long)row * ldc + col;
        if (Cf) Cf[idx] = v;
        if (Ch) Ch[idx] = (_Float16)v;
      }
    }
  }
}

// ---------------------------------------------------------------------------
// Elementwise f32 -> f16
// ---------------------------------------------------------------------------
__global__ void f32_to_f16(const float* __restrict__ src, _Float16* __restrict__ dst, long n) {
  long i = (long)blockIdx.x * blockDim.x + threadIdx.x;
  if (i < n) dst[i] = (_Float16)src[i];
}

// ---------------------------------------------------------------------------
// Repack qkv (16384 x 1536, f16) into q[b,h,s,d], k[b,h,s,d], vT[b,h,d,s]
// ---------------------------------------------------------------------------
__global__ void repack_qkv(const _Float16* __restrict__ qkv,
                           _Float16* __restrict__ q, _Float16* __restrict__ k,
                           _Float16* __restrict__ vt) {
  long i = (long)blockIdx.x * blockDim.x + threadIdx.x;
  if (i >= 16384L * 1536L) return;
  int n = (int)(i % 1536);
  long m = i / 1536;
  int b = (int)(m / 512), s = (int)(m % 512);
  int which = n / 512;
  int e = n % 512;
  int h = e / 256, d = e % 256;
  _Float16 val = qkv[i];
  long zi = (long)(b * 2 + h);
  if (which == 0)      q[(zi * 512 + s) * 256 + d] = val;
  else if (which == 1) k[(zi * 512 + s) * 256 + d] = val;
  else                 vt[(zi * 256 + d) * 512 + s] = val;
}

// ---------------------------------------------------------------------------
// Row softmax: f32 scores (512 cols) -> f16 probabilities
// ---------------------------------------------------------------------------
__global__ void softmax_rows(const float* __restrict__ S, _Float16* __restrict__ P, int ncol) {
  __shared__ float red[256];
  const long row = blockIdx.x;
  const float* src = S + row * (long)ncol;
  _Float16* dst = P + row * (long)ncol;

  float mx = -1e30f;
  for (int c = threadIdx.x; c < ncol; c += blockDim.x) mx = fmaxf(mx, src[c]);
  red[threadIdx.x] = mx;
  __syncthreads();
  for (int s = blockDim.x / 2; s > 0; s >>= 1) {
    if (threadIdx.x < s) red[threadIdx.x] = fmaxf(red[threadIdx.x], red[threadIdx.x + s]);
    __syncthreads();
  }
  mx = red[0];
  __syncthreads();

  float sum = 0.f;
  for (int c = threadIdx.x; c < ncol; c += blockDim.x) sum += __expf(src[c] - mx);
  red[threadIdx.x] = sum;
  __syncthreads();
  for (int s = blockDim.x / 2; s > 0; s >>= 1) {
    if (threadIdx.x < s) red[threadIdx.x] += red[threadIdx.x + s];
    __syncthreads();
  }
  const float inv = 1.f / red[0];
  for (int c = threadIdx.x; c < ncol; c += blockDim.x)
    dst[c] = (_Float16)(__expf(src[c] - mx) * inv);
}

// ---------------------------------------------------------------------------
// Emissions (dec @ crf_w^T + crf_b) and seg log_softmax (dec @ ent_w^T + ent_b)
// One wave per token; dec row staged in LDS.
// ---------------------------------------------------------------------------
__global__ __launch_bounds__(32)
void emis_seg(const float* __restrict__ dec,
              const float* __restrict__ crf_w, const float* __restrict__ crf_b,
              const float* __restrict__ ent_w, const float* __restrict__ ent_b,
              float* __restrict__ em, float* __restrict__ seg) {
  __shared__ float x[512];
  __shared__ float sv[2];
  const long m = blockIdx.x;
  const float* row = dec + m * 512;
  for (int c = threadIdx.x; c < 512; c += 32) x[c] = row[c];
  __syncthreads();

  const int t = threadIdx.x;
  if (t < 24) {
    float acc = crf_b[t];
    const float* w = crf_w + t * 512;
    for (int k = 0; k < 512; ++k) acc += x[k] * w[k];
    em[m * 24 + t] = acc;
  } else if (t < 26) {
    const int c = t - 24;
    float acc = ent_b[c];
    const float* w = ent_w + c * 512;
    for (int k = 0; k < 512; ++k) acc += x[k] * w[k];
    sv[c] = acc;
  }
  __syncthreads();
  if (t == 0) {
    const float s0 = sv[0], s1 = sv[1];
    const float mx = fmaxf(s0, s1);
    const float lse = mx + __logf(__expf(s0 - mx) + __expf(s1 - mx));
    seg[m * 2 + 0] = s0 - lse;
    seg[m * 2 + 1] = s1 - lse;
  }
}

// ---------------------------------------------------------------------------
// CRF forward: normalizer (lanes = tags) + numerator (lane 0), one wave/batch
// ---------------------------------------------------------------------------
__global__ __launch_bounds__(32)
void crf_forward(const float* __restrict__ em, const int* __restrict__ labels,
                 const int* __restrict__ mask,
                 const float* __restrict__ start_t, const float* __restrict__ end_t,
                 const float* __restrict__ trans, float* __restrict__ numden) {
  __shared__ float tr[576];
  __shared__ float sc[24], ns[24];
  const int b = blockIdx.x, j = threadIdx.x;
  for (int i = j; i < 576; i += 32) tr[i] = trans[i];
  const float* e = em + (long)b * 512 * 24;
  if (j < 24) sc[j] = start_t[j] + e[j];
  __syncthreads();

  for (int t = 1; t < 512; ++t) {
    const int mt = mask[b * 512 + t];
    if (j < 24) {
      float mx = -1e30f;
      for (int i = 0; i < 24; ++i) mx = fmaxf(mx, sc[i] + tr[i * 24 + j]);
      float s = 0.f;
      for (int i = 0; i < 24; ++i) s += __expf(sc[i] + tr[i * 24 + j] - mx);
      const float nxt = mx + __logf(s) + e[t * 24 + j];
      ns[j] = (mt != 0) ? nxt : sc[j];
    }
    __syncthreads();
    if (j < 24) sc[j] = ns[j];
    __syncthreads();
  }

  if (j == 0) {
    float mx = -1e30f;
    for (int i = 0; i < 24; ++i) mx = fmaxf(mx, sc[i] + end_t[i]);
    float s = 0.f;
    for (int i = 0; i < 24; ++i) s += __expf(sc[i] + end_t[i] - mx);
    const float den = mx + __logf(s);

    const int* lab = labels + b * 512;
    const int* mk  = mask + b * 512;
    float num = start_t[lab[0]] + e[lab[0]];
    for (int t = 1; t < 512; ++t)
      if (mk[t]) num += trans[lab[t - 1] * 24 + lab[t]] + e[t * 24 + lab[t]];
    int cnt = 0;
    for (int t = 0; t < 512; ++t) cnt += (mk[t] != 0);
    num += end_t[lab[cnt - 1]];

    numden[b * 2 + 0] = num;
    numden[b * 2 + 1] = den;
  }
}

// ---------------------------------------------------------------------------
// Viterbi: forward max/argmax with history, then lane-0 backtrack
// ---------------------------------------------------------------------------
__global__ __launch_bounds__(32)
void crf_viterbi(const float* __restrict__ em,
                 const float* __restrict__ start_t, const float* __restrict__ end_t,
                 const float* __restrict__ trans,
                 int* __restrict__ hist, float* __restrict__ crf_out) {
  __shared__ float tr[576];
  __shared__ float sc[24], ns[24];
  const int b = blockIdx.x, j = threadIdx.x;
  for (int i = j; i < 576; i += 32) tr[i] = trans[i];
  const float* e = em + (long)b * 512 * 24;
  if (j < 24) sc[j] = start_t[j] + e[j];
  __syncthreads();

  for (int t = 1; t < 512; ++t) {
    if (j < 24) {
      float best = -1e30f;
      int bi = 0;
      for (int i = 0; i < 24; ++i) {
        const float v = sc[i] + tr[i * 24 + j];
        if (v > best) { best = v; bi = i; }
      }
      ns[j] = best + e[t * 24 + j];
      hist[((long)(t - 1) * 32 + b) * 24 + j] = bi;
    }
    __syncthreads();
    if (j < 24) sc[j] = ns[j];
    __syncthreads();
  }

  if (j == 0) {
    int tag = 0;
    float best = -1e30f;
    for (int i = 0; i < 24; ++i) {
      const float v = sc[i] + end_t[i];
      if (v > best) { best = v; tag = i; }
    }
    crf_out[b * 512 + 511] = (float)tag;
    for (int t = 510; t >= 0; --t) {
      tag = hist[((long)t * 32 + b) * 24 + tag];
      crf_out[b * 512 + t] = (float)tag;
    }
  }
}

// ---------------------------------------------------------------------------
// -llh = -(sum(num - den) / sum(mask))
// ---------------------------------------------------------------------------
__global__ void finalize_loss(const float* __restrict__ numden,
                              const int* __restrict__ mask, float* __restrict__ out) {
  __shared__ float r1[256], r2[256];
  float s = 0.f, ms = 0.f;
  for (int i = threadIdx.x; i < 32; i += 256) s += numden[2 * i] - numden[2 * i + 1];
  for (int i = threadIdx.x; i < 16384; i += 256) ms += (mask[i] != 0) ? 1.f : 0.f;
  r1[threadIdx.x] = s;
  r2[threadIdx.x] = ms;
  __syncthreads();
  for (int st = 128; st > 0; st >>= 1) {
    if (threadIdx.x < st) {
      r1[threadIdx.x] += r1[threadIdx.x + st];
      r2[threadIdx.x] += r2[threadIdx.x + st];
    }
    __syncthreads();
  }
  if (threadIdx.x == 0) out[0] = -(r1[0] / r2[0]);
}

// ---------------------------------------------------------------------------
extern "C" void kernel_launch(void* const* d_in, const int* in_sizes, int n_in,
                              void* d_out, int out_size, void* d_ws, size_t ws_size,
                              hipStream_t stream) {
  const float* X       = (const float*)d_in[0];
  const int*   labels  = (const int*)d_in[1];
  const int*   mask    = (const int*)d_in[2];
  const float* Win     = (const float*)d_in[3];
  const float* bin     = (const float*)d_in[4];
  const float* Wout    = (const float*)d_in[5];
  const float* bout    = (const float*)d_in[6];
  const float* crf_w   = (const float*)d_in[7];
  const float* crf_b   = (const float*)d_in[8];
  const float* start_t = (const float*)d_in[9];
  const float* end_t   = (const float*)d_in[10];
  const float* trans   = (const float*)d_in[11];
  const float* ent_w   = (const float*)d_in[12];
  const float* ent_b   = (const float*)d_in[13];
  float* out = (float*)d_out;

  char* ws = (char*)d_ws;
  size_t off = 0;
  auto take = [&](size_t bytes) -> char* {
    char* p = ws + off;
    off += (bytes + 255) & ~(size_t)255;
    return p;
  };
  _Float16* Xh     = (_Float16*)take(16384UL * 512 * 2);
  _Float16* Winh   = (_Float16*)take(1536UL * 512 * 2);
  _Float16* Wouth  = (_Float16*)take(512UL * 512 * 2);
  _Float16* qkvh   = (_Float16*)take(16384UL * 1536 * 2);
  _Float16* qh     = (_Float16*)take(64UL * 512 * 256 * 2);
  _Float16* kh     = (_Float16*)take(64UL * 512 * 256 * 2);
  _Float16* vt     = (_Float16*)take(64UL * 256 * 512 * 2);
  float*    scores = (float*)take(64UL * 512 * 512 * 4);
  _Float16* attnh  = (_Float16*)take(64UL * 512 * 512 * 2);
  _Float16* aouth  = (_Float16*)take(16384UL * 512 * 2);
  float*    dec    = (float*)take(16384UL * 512 * 4);
  float*    em     = (float*)take(16384UL * 24 * 4);
  int*      hist   = (int*)take(511UL * 32 * 24 * 4);
  float*    numden = (float*)take(64 * 4);

  // f16 conversions
  f32_to_f16<<<dim3(32768), 256, 0, stream>>>(X, Xh, 8388608L);
  f32_to_f16<<<dim3(3072), 256, 0, stream>>>(Win, Winh, 786432L);
  f32_to_f16<<<dim3(1024), 256, 0, stream>>>(Wout, Wouth, 262144L);

  // QKV projection: (16384x512) @ (1536x512)^T + bin -> f16
  wmma_gemm_abt<<<dim3(24, 256, 1), 32, 0, stream>>>(
      Xh, 512, 0, Winh, 512, 0, nullptr, qkvh, 1536, 0, 0, 1, bin, 1.f, 0, 512);

  repack_qkv<<<dim3(98304), 256, 0, stream>>>(qkvh, qh, kh, vt);

  // scores = q @ k^T / sqrt(256), per (b,h): 512x512x256 -> f32
  wmma_gemm_abt<<<dim3(8, 8, 64), 32, 0, stream>>>(
      qh, 256, 512L * 256, kh, 256, 512L * 256, scores, nullptr, 512,
      512L * 512, 0, 1, nullptr, 0.0625f, 0, 256);

  softmax_rows<<<dim3(32768), 256, 0, stream>>>(scores, attnh, 512);

  // out = attn @ v, per (b,h): 512x256x512; scatter to token-major f16
  wmma_gemm_abt<<<dim3(4, 8, 64), 32, 0, stream>>>(
      attnh, 512, 512L * 512, vt, 512, 256L * 512, nullptr, aouth, 512,
      512L * 512, 256, 2, nullptr, 1.f, 0, 512);

  // output projection + bias + relu -> dec (f32)
  wmma_gemm_abt<<<dim3(8, 256, 1), 32, 0, stream>>>(
      aouth, 512, 0, Wouth, 512, 0, dec, nullptr, 512, 0, 0, 1, bout, 1.f, 1, 512);

  emis_seg<<<dim3(16384), 32, 0, stream>>>(dec, crf_w, crf_b, ent_w, ent_b, em, out + 16384);
  crf_forward<<<dim3(32), 32, 0, stream>>>(em, labels, mask, start_t, end_t, trans, numden);
  crf_viterbi<<<dim3(32), 32, 0, stream>>>(em, start_t, end_t, trans, hist, out);
  finalize_loss<<<dim3(1), 256, 0, stream>>>(numden, mask, out + 16384 + 32768);
}
